// DAttentionBase_20435454395185
// MI455X (gfx1250) — matrix operand order repaired and checked by reading, not compiled, not run on hardware
//
#include <hip/hip_runtime.h>
#include <hip/hip_bf16.h>

// ---------------------------------------------------------------------------
// Deformable attention (DAttentionBase) for MI455X / gfx1250, wave32 + WMMA.
//   B=8 C=256 H=W=32 G=4 NH=8 HC=32 GC=64 ns=1024
// GEMMs: v_wmma_f32_16x16x32_bf16 (f32 accumulate). Attention fully fused
// per 16-query tile (attn matrix never touches memory). Per-head k^T and v
// operands are staged in LDS via the CDNA5 Tensor Data Mover
// (tensor_load_to_lds + s_wait_tensorcnt) with a cooperative-copy fallback.
// ---------------------------------------------------------------------------

typedef __bf16 bf16_t;
typedef __attribute__((ext_vector_type(16))) __bf16 v16bf;
typedef __attribute__((ext_vector_type(8)))  float  v8f;
typedef unsigned int u32x4 __attribute__((ext_vector_type(4)));
typedef int          i32x4 __attribute__((ext_vector_type(4)));
typedef int          i32x8 __attribute__((ext_vector_type(8)));

#define BB   8
#define CCH  256
#define HWD  1024
#define GG   4
#define NHH  8
#define HCC  32
#define GCC  64
#define NSS  1024
#define RPEW 63
#define SCALE_F 0.17677669529663689f   // 32^-0.5

static __device__ __forceinline__ bf16_t f2bf(float x) { return (bf16_t)x; }

// --------------------------- TDM (Tensor Data Mover) ------------------------
#if __has_builtin(__builtin_amdgcn_tensor_load_to_lds)
#define HAVE_TDM 1
#if __has_include(<hip/amd_detail/amd_gfx1250_TDM.h>)
// therock-10.0 headers -> 6-arg builtin
#define TDM_ISSUE(g0, g1) \
  __builtin_amdgcn_tensor_load_to_lds((g0), (g1), (i32x4)(0), (i32x4)(0), (i32x8)(0), 0)
#else
// ROCm 7.2 -> 5-arg builtin
#define TDM_ISSUE(g0, g1) \
  __builtin_amdgcn_tensor_load_to_lds((g0), (g1), (i32x4)(0), (i32x4)(0), 0)
#endif
#else
#define HAVE_TDM 0
#endif

// Generic LDS pointer -> 32-bit LDS byte offset (aperture low bits).
static __device__ __forceinline__ unsigned ldsOffset(const void* p) {
  return (unsigned)(unsigned long long)p;
}

#if HAVE_TDM
// 2D tile of 2-byte elements: tile_d0 contiguous elems per row, tile_d1 rows,
// row stride stride0 (elements). LDS receives rows packed contiguously.
static __device__ __forceinline__ void tdm_load_2d_b16(
    unsigned lds_off, const void* gptr, unsigned tile_d0, unsigned tile_d1,
    unsigned long long stride0)
{
  const unsigned long long ga = (unsigned long long)gptr;
  u32x4 g0;
  g0[0] = 1u;                                     // count=1, user mode
  g0[1] = lds_off;                                // lds_addr
  g0[2] = (unsigned)ga;                           // global_addr[31:0]
  g0[3] = (unsigned)((ga >> 32) & 0x01FFFFFFu) | 0x80000000u;  // [56:32] | type=2
  const unsigned dim0 = 0x7FFFFFFFu, dim1 = 0x7FFFFFFFu;       // no OOB clip
  i32x8 g1;
  g1[0] = (int)(1u << 16);                        // wg_mask=0, data_size=1 (2B)
  g1[1] = (int)((dim0 & 0xFFFFu) << 16);          // tensor_dim0[15:0]
  g1[2] = (int)(((dim0 >> 16) & 0xFFFFu) | ((dim1 & 0xFFFFu) << 16));
  g1[3] = (int)(((dim1 >> 16) & 0xFFFFu) | (tile_d0 << 16));   // tile_dim0
  g1[4] = (int)(tile_d1 & 0xFFFFu);               // tile_dim1, tile_dim2=0
  g1[5] = (int)(unsigned)(stride0 & 0xFFFFFFFFull);            // dim0_stride lo
  g1[6] = (int)(unsigned)((stride0 >> 32) & 0xFFFFull);        // hi16, dim1_stride=0
  g1[7] = 0;
  TDM_ISSUE(g0, g1);
}
#endif

// ---------------------------------------------------------------------------
// conv1x1 as GEMM: acc[m,n] = bias[m] + sum_k Wt[m,k] * X[b][k,n]
// Compile-time output modes: OUTF -> f32 (M,N); OUT16 -> bf16,
// TRANS ? (N,M) : (M,N). Branch-free epilogue per instantiation.
// One wave = one 16x16 D tile, bf16 WMMA per 32-wide K step.
// ---------------------------------------------------------------------------
template <int OUTF, int OUT16, int TRANS>
__global__ void __launch_bounds__(256) k_conv1x1(
    const float* __restrict__ Wt, const float* __restrict__ bias,
    const float* __restrict__ X, float* __restrict__ Yf,
    bf16_t* __restrict__ Y16,
    int M, int K, int N, int batches)
{
  const int lane  = threadIdx.x & 31;
  const int wave  = blockIdx.x * (blockDim.x >> 5) + (threadIdx.x >> 5);
  const int ntiles = N >> 4;
  const int tilesPerBatch = (M >> 4) * ntiles;
  const int b = wave / tilesPerBatch;
  if (b >= batches) return;
  const int t  = wave % tilesPerBatch;
  const int mb = (t / ntiles) << 4;
  const int nb = (t % ntiles) << 4;

  const float* Xb = X + (size_t)b * K * N;

  const int mA    = mb + (lane & 15);
  const int nB    = nb + (lane & 15);
  const int kaOff = (lane >> 4) << 3;
  const int kbOff = (lane >> 4) << 3;

  v8f acc = {};
  for (int kk = 0; kk < K; kk += 32) {
    v16bf a, bf;
    const float* Arow = Wt + (size_t)mA * K + kk + kaOff;
#pragma unroll
    for (int i = 0; i < 8; ++i) { a[i] = f2bf(Arow[i]); a[i + 8] = f2bf(Arow[16 + i]); }
    const float* Bcol = Xb + (size_t)(kk + kbOff) * N + nB;
#pragma unroll
    for (int i = 0; i < 8; ++i) { bf[i] = f2bf(Bcol[(size_t)i * N]); bf[i + 8] = f2bf(Bcol[(size_t)(16 + i) * N]); }
    acc = __builtin_amdgcn_wmma_f32_16x16x32_bf16(false, a, false, bf, (short)0, acc,
                                                  false, false);
  }
  const int mRow = mb + ((lane >> 4) << 3);
  const int nCol = nb + (lane & 15);
#pragma unroll
  for (int r = 0; r < 8; ++r) {
    const float val = acc[r] + bias[mRow + r];
    if constexpr (OUTF)
      Yf[(size_t)b * M * N + (size_t)(mRow + r) * N + nCol] = val;
    if constexpr (OUT16) {
      if constexpr (TRANS)
        Y16[(size_t)b * M * N + (size_t)nCol * M + (mRow + r)] = f2bf(val);
      else
        Y16[(size_t)b * M * N + (size_t)(mRow + r) * N + nCol] = f2bf(val);
    }
  }
}

// ---------------------------------------------------------------------------
// Offset network: depthwise 5x5 (pad 2) -> LayerNorm(ch) -> GELU(exact) ->
// 1x1 (GC->2) -> +ref -> tanh. One thread per (bg,h,w); mean/var recompute.
// ---------------------------------------------------------------------------
static __device__ __forceinline__ float dwconv_at(
    const float* __restrict__ qg, const float* __restrict__ dww, float dwb,
    int h, int w)
{
  float acc = dwb;
#pragma unroll
  for (int ky = 0; ky < 5; ++ky) {
    const int yy = h + ky - 2;
    if ((unsigned)yy >= 32u) continue;
#pragma unroll
    for (int kx = 0; kx < 5; ++kx) {
      const int xx = w + kx - 2;
      if ((unsigned)xx >= 32u) continue;
      acc += qg[yy * 32 + xx] * dww[ky * 5 + kx];
    }
  }
  return acc;
}

__global__ void __launch_bounds__(128) k_offset(
    const float* __restrict__ q, const float* __restrict__ dww,
    const float* __restrict__ dwb, const float* __restrict__ lng,
    const float* __restrict__ lnb, const float* __restrict__ pw,
    float* __restrict__ pos, float* __restrict__ pos_out, float* __restrict__ ref_out)
{
  const int tid = blockIdx.x * blockDim.x + threadIdx.x;
  if (tid >= BB * GG * HWD) return;
  const int bg = tid / HWD, p = tid % HWD;
  const int h = p >> 5, w = p & 31;
  const int b = bg >> 2, g = bg & 3;
  const float* qbase = q + ((size_t)b * CCH + g * GCC) * HWD;

  float s = 0.f, s2 = 0.f;
  for (int c = 0; c < GCC; ++c) {
    const float v = dwconv_at(qbase + (size_t)c * HWD, dww + c * 25, dwb[c], h, w);
    s += v; s2 += v * v;
  }
  const float mu   = s * (1.f / GCC);
  const float var  = s2 * (1.f / GCC) - mu * mu;
  const float rstd = rsqrtf(var + 1e-5f);

  float o0 = 0.f, o1 = 0.f;
  for (int c = 0; c < GCC; ++c) {
    const float v  = dwconv_at(qbase + (size_t)c * HWD, dww + c * 25, dwb[c], h, w);
    const float xn = (v - mu) * rstd * lng[c] + lnb[c];
    const float ge = 0.5f * xn * (1.f + erff(xn * 0.70710678118654752f));
    o0 += ge * pw[c];
    o1 += ge * pw[GCC + c];
  }
  const float ry = ((float)h + 0.5f) * (2.f / 32.f) - 1.f;
  const float rx = ((float)w + 0.5f) * (2.f / 32.f) - 1.f;
  const float p0 = tanhf(o0 + ry);
  const float p1 = tanhf(o1 + rx);
  pos[(size_t)tid * 2]         = p0; pos[(size_t)tid * 2 + 1]     = p1;
  pos_out[(size_t)tid * 2]     = p0; pos_out[(size_t)tid * 2 + 1] = p1;
  ref_out[(size_t)tid * 2]     = ry; ref_out[(size_t)tid * 2 + 1] = rx;
}

// ---------------------------------------------------------------------------
// Bilinear grid-sample of x at pos (align_corners=True, zeros padding).
// Faithful bug: pos[...,0] (y-channel) is used as the x (width) coordinate.
// ---------------------------------------------------------------------------
__global__ void __launch_bounds__(128) k_sample(
    const float* __restrict__ x, const float* __restrict__ pos,
    float* __restrict__ xs)
{
  const int tid = blockIdx.x * blockDim.x + threadIdx.x;
  if (tid >= BB * GG * NSS) return;
  const int bg = tid / NSS, n = tid % NSS;
  const int b = bg >> 2, g = bg & 3;

  const float gx = pos[(size_t)tid * 2];
  const float gy = pos[(size_t)tid * 2 + 1];
  const float fx = (gx + 1.f) * 0.5f * 31.f;
  const float fy = (gy + 1.f) * 0.5f * 31.f;
  const float x0f = floorf(fx), y0f = floorf(fy);
  const int x0 = (int)x0f, y0 = (int)y0f, x1 = x0 + 1, y1 = y0 + 1;
  const float wx1 = fx - x0f, wy1 = fy - y0f;
  const float wx0 = 1.f - wx1, wy0 = 1.f - wy1;
  const bool vx0 = (unsigned)x0 < 32u, vx1 = (unsigned)x1 < 32u;
  const bool vy0 = (unsigned)y0 < 32u, vy1 = (unsigned)y1 < 32u;
  const int cx0 = min(max(x0, 0), 31), cx1 = min(max(x1, 0), 31);
  const int cy0 = min(max(y0, 0), 31), cy1 = min(max(y1, 0), 31);

  const float* img = x  + ((size_t)b * CCH + g * GCC) * HWD;
  float*       dst = xs + ((size_t)b * CCH + g * GCC) * NSS + n;
  for (int c = 0; c < GCC; ++c, img += HWD, dst += NSS) {
    const float v00 = (vx0 && vy0) ? img[cy0 * 32 + cx0] : 0.f;
    const float v01 = (vx1 && vy0) ? img[cy0 * 32 + cx1] : 0.f;
    const float v10 = (vx0 && vy1) ? img[cy1 * 32 + cx0] : 0.f;
    const float v11 = (vx1 && vy1) ? img[cy1 * 32 + cx1] : 0.f;
    *dst = v00 * (wx0 * wy0) + v01 * (wx1 * wy0) + v10 * (wx0 * wy1) + v11 * (wx1 * wy1);
  }
}

// ---------------------------------------------------------------------------
// Fused attention. 64 blocks (one head each) x 512 threads (16 waves).
// TDM stages this head's k^T (1024x32 bf16) and v (32x1024 bf16) in LDS once;
// rpe (63x63 f32) and pos (1024x2 f32) also LDS-resident. Each wave owns 4
// query tiles; per 32-key chunk: 2 WMMA attn, bias from LDS rpe, bf16 round,
// LDS transpose to A-layout (s_wait_dscnt), 2 WMMA accumulating out @ v^T.
// Dynamic LDS: 172 KB (one workgroup per WGP; well under 320 KB).
// ---------------------------------------------------------------------------
#define SM_RPE   0
#define SM_POS   16000
#define SM_KBUF  24192
#define SM_VBUF  (24192 + 65536)
#define SM_STAGE (24192 + 131072)
#define SM_TOTAL (SM_STAGE + 16 * 512 * 2)

static __device__ __forceinline__ float bilinear63(const float* __restrict__ img,
                                                   float gx, float gy)
{
  const float fx = (gx + 1.f) * 0.5f * 62.f;
  const float fy = (gy + 1.f) * 0.5f * 62.f;
  const float x0f = floorf(fx), y0f = floorf(fy);
  const int x0 = (int)x0f, y0 = (int)y0f, x1 = x0 + 1, y1 = y0 + 1;
  const float wx1 = fx - x0f, wy1 = fy - y0f;
  const float wx0 = 1.f - wx1, wy0 = 1.f - wy1;
  const bool vx0 = (unsigned)x0 < 63u, vx1 = (unsigned)x1 < 63u;
  const bool vy0 = (unsigned)y0 < 63u, vy1 = (unsigned)y1 < 63u;
  const int cx0 = min(max(x0, 0), 62), cx1 = min(max(x1, 0), 62);
  const int cy0 = min(max(y0, 0), 62), cy1 = min(max(y1, 0), 62);
  const float v00 = (vx0 && vy0) ? img[cy0 * 63 + cx0] : 0.f;
  const float v01 = (vx1 && vy0) ? img[cy0 * 63 + cx1] : 0.f;
  const float v10 = (vx0 && vy1) ? img[cy1 * 63 + cx0] : 0.f;
  const float v11 = (vx1 && vy1) ? img[cy1 * 63 + cx1] : 0.f;
  return v00 * (wx0 * wy0) + v01 * (wx1 * wy0) + v10 * (wx0 * wy1) + v11 * (wx1 * wy1);
}

__global__ void __launch_bounds__(512) k_attn(
    const bf16_t* __restrict__ qT, const bf16_t* __restrict__ kT,
    const bf16_t* __restrict__ v16, const float* __restrict__ pos,
    const float* __restrict__ rpe, float* __restrict__ outw)
{
  extern __shared__ __align__(64) char smem[];
  float*  rpes  = (float*)(smem + SM_RPE);
  float*  posb  = (float*)(smem + SM_POS);
  bf16_t* kbuf  = (bf16_t*)(smem + SM_KBUF);     // [n][c] 1024x32
  bf16_t* vbuf  = (bf16_t*)(smem + SM_VBUF);     // [c][n] 32x1024
  bf16_t* stage = (bf16_t*)(smem + SM_STAGE);    // per-wave 16x32

  const int head = blockIdx.x;
  const int b    = head >> 3;
  const int nh   = head & 7;
  const int ch0  = nh * HCC;
  const int wv   = threadIdx.x >> 5;
  const int lane = threadIdx.x & 31;

  const bf16_t* kT_head = kT  + (size_t)b * NSS * CCH + ch0;   // row stride CCH
  const bf16_t* v_head  = v16 + (size_t)head * (HCC * NSS);    // contiguous
  const bf16_t* qT_head = qT  + (size_t)b * HWD * CCH + ch0;   // row stride CCH
  const float*  ph      = pos + (size_t)(head >> 1) * (NSS * 2);
  const float*  rh      = rpe + (size_t)nh * (RPEW * RPEW);

  // --- stage per-head operands in LDS --------------------------------------
  for (int i = threadIdx.x; i < RPEW * RPEW; i += 512) rpes[i] = rh[i];
  for (int i = threadIdx.x; i < NSS * 2; i += 512)     posb[i] = ph[i];
#if HAVE_TDM
  if (threadIdx.x < 32) {
    tdm_load_2d_b16(ldsOffset(kbuf), kT_head, HCC, NSS, (unsigned long long)CCH);
    tdm_load_2d_b16(ldsOffset(vbuf), v_head, HCC * NSS, 1, (unsigned long long)(HCC * NSS));
#if __has_builtin(__builtin_amdgcn_s_wait_tensorcnt)
    __builtin_amdgcn_s_wait_tensorcnt(0);
#else
    asm volatile("s_wait_tensorcnt 0" ::: "memory");
#endif
  }
#else
  for (int i = threadIdx.x; i < HCC * NSS; i += 512) {
    kbuf[i] = kT_head[(size_t)(i >> 5) * CCH + (i & 31)];
    vbuf[i] = v_head[i];
  }
#endif
  __syncthreads();

  bf16_t* stg = stage + wv * 512;
  const int mRowL = (lane >> 4) << 3;   // D-layout row base within tile
  const int nOff  = lane & 15;          // D-layout column within tile

  for (int mt = 0; mt < 4; ++mt) {
    const int mb = ((wv << 2) + mt) << 4;

    // q fragment (A-layout, K = HC = 32), contiguous bf16 rows of qT
    v16bf a1;
    {
      const bf16_t* qrow = qT_head + (size_t)(mb + (lane & 15)) * CCH + ((lane >> 4) << 3);
#pragma unroll
      for (int i = 0; i < 8; ++i) { a1[i] = qrow[i]; a1[i + 8] = qrow[16 + i]; }
    }

    v8f accO0 = {}, accO1 = {};
    const int mRow = mb + mRowL;

    for (int nc = 0; nc < NSS; nc += 32) {
#pragma unroll
      for (int t = 0; t < 2; ++t) {
        const int nt = nc + (t << 4);
        // k fragment (B-layout 32x16): two 16B LDS runs of kbuf (k = 8h+i, 16+8h+i)
        v16bf b1;
        {
          const bf16_t* krow = kbuf + (size_t)(nt + (lane & 15)) * HCC + ((lane >> 4) << 3);
#pragma unroll
          for (int i = 0; i < 8; ++i) { b1[i] = krow[i]; b1[i + 8] = krow[16 + i]; }
        }
        v8f at = {};
        at = __builtin_amdgcn_wmma_f32_16x16x32_bf16(false, a1, false, b1, (short)0, at,
                                                     false, false);
        const int   n_e = nt + nOff;
        const float py  = posb[2 * n_e];
        const float px  = posb[2 * n_e + 1];
#pragma unroll
        for (int r = 0; r < 8; ++r) {
          const int   m_e = mRow + r;
          const float qgy = ((float)((m_e >> 5) & 31) + 0.5f) * (2.f / 32.f) - 1.f;
          const float qgx = ((float)(m_e & 31) + 0.5f) * (2.f / 32.f) - 1.f;
          const float bias = bilinear63(rpes, 0.5f * (qgx - px), 0.5f * (qgy - py));
          stg[(mRowL + r) * 32 + (t << 4) + nOff] = f2bf(at[r] * SCALE_F + bias);
        }
      }
      asm volatile("s_wait_dscnt 0" ::: "memory");   // LDS RAW fence (wave-local)

      // attn tile in A-layout (K = key index within chunk)
      v16bf a2;
      {
        const int m  = lane & 15;
        const int kb = (lane >> 4) << 3;
#pragma unroll
        for (int i = 0; i < 8; ++i) {
          a2[i]     = stg[m * 32 + kb + i];
          a2[i + 8] = stg[m * 32 + kb + 16 + i];
        }
      }
      // v fragments (B-layout: K = key, N = channel) from LDS
#pragma unroll
      for (int ct = 0; ct < 2; ++ct) {
        v16bf b2;
        const bf16_t* vrow = vbuf + (size_t)((ct << 4) + (lane & 15)) * NSS
                                  + nc + ((lane >> 4) << 3);
#pragma unroll
        for (int i = 0; i < 8; ++i) { b2[i] = vrow[i]; b2[i + 8] = vrow[16 + i]; }
        if (ct == 0)
          accO0 = __builtin_amdgcn_wmma_f32_16x16x32_bf16(false, a2, false, b2, (short)0,
                                                          accO0, false, false);
        else
          accO1 = __builtin_amdgcn_wmma_f32_16x16x32_bf16(false, a2, false, b2, (short)0,
                                                          accO1, false, false);
      }
    }

    float* oh = outw + (size_t)head * (HCC * HWD);
#pragma unroll
    for (int r = 0; r < 8; ++r) {
      oh[(size_t)nOff * HWD        + (mRow + r)] = accO0[r];
      oh[(size_t)(16 + nOff) * HWD + (mRow + r)] = accO1[r];
    }
  }
}

// ---------------------------------------------------------------------------
extern "C" void kernel_launch(void* const* d_in, const int* in_sizes, int n_in,
                              void* d_out, int out_size, void* d_ws, size_t ws_size,
                              hipStream_t stream)
{
  (void)in_sizes; (void)n_in; (void)out_size; (void)ws_size;
  const float* x   = (const float*)d_in[0];
  const float* wq  = (const float*)d_in[1];
  const float* bq  = (const float*)d_in[2];
  const float* wk  = (const float*)d_in[3];
  const float* bk  = (const float*)d_in[4];
  const float* wv  = (const float*)d_in[5];
  const float* bv  = (const float*)d_in[6];
  const float* wo  = (const float*)d_in[7];
  const float* bo  = (const float*)d_in[8];
  const float* dww = (const float*)d_in[9];
  const float* dwb = (const float*)d_in[10];
  const float* lng = (const float*)d_in[11];
  const float* lnb = (const float*)d_in[12];
  const float* pw  = (const float*)d_in[13];
  const float* rpe = (const float*)d_in[14];

  float* y_out   = (float*)d_out;                                  // (B,C,H,W)
  float* pos_out = y_out + (size_t)BB * CCH * HWD;                 // (B,G,32,32,2)
  float* ref_out = pos_out + (size_t)BB * GG * HWD * 2;            // (B,G,32,32,2)

  float* ws     = (float*)d_ws;
  float* q_ws   = ws;  ws += (size_t)BB * CCH * HWD;       // f32 q (offset net)
  float* xs_ws  = ws;  ws += (size_t)BB * CCH * NSS;       // f32 x_sampled
  float* out_ws = ws;  ws += (size_t)BB * CCH * HWD;       // f32 attn out
  float* pos_ws = ws;  ws += (size_t)BB * GG * NSS * 2;    // f32 pos
  bf16_t* qT16  = (bf16_t*)ws;  ws += (size_t)BB * CCH * HWD / 2;  // bf16 qT (HW,C)
  bf16_t* kT16  = (bf16_t*)ws;  ws += (size_t)BB * CCH * NSS / 2;  // bf16 kT (ns,C)
  bf16_t* v16   = (bf16_t*)ws;  ws += (size_t)BB * CCH * NSS / 2;  // bf16 v  (C,ns)

  // 1) q projection: f32 (C,HW) for offset net + bf16 transposed for attention
  k_conv1x1<1, 1, 1><<<1024, 256, 0, stream>>>(wq, bq, x, q_ws, qT16,
                                               CCH, CCH, HWD, BB);
  // 2) offset network -> pos (+ pos/ref outputs)
  k_offset<<<256, 128, 0, stream>>>(q_ws, dww, dwb, lng, lnb, pw,
                                    pos_ws, pos_out, ref_out);
  // 3) x_sampled = grid_sample(x, pos)
  k_sample<<<256, 128, 0, stream>>>(x, pos_ws, xs_ws);
  // 4) k (bf16 transposed), v (bf16) projections of x_sampled
  k_conv1x1<0, 1, 1><<<1024, 256, 0, stream>>>(wk, bk, xs_ws, (float*)nullptr, kT16,
                                               CCH, CCH, NSS, BB);
  k_conv1x1<0, 1, 0><<<1024, 256, 0, stream>>>(wv, bv, xs_ws, (float*)nullptr, v16,
                                               CCH, CCH, NSS, BB);
  // 5) fused (q^T k * scale + rpe bias) @ v^T, TDM-staged operands
  k_attn<<<64, 512, SM_TOTAL, stream>>>(qT16, kT16, v16, pos_ws, rpe, out_ws);
  // 6) y = wo @ out + bo
  k_conv1x1<1, 0, 0><<<1024, 256, 0, stream>>>(wo, bo, out_ws, y_out, (bf16_t*)nullptr,
                                               CCH, CCH, HWD, BB);
}
